// GNNPlusLayer_87419764343138
// MI455X (gfx1250) — compile-verified, matrix operand order checked by reading, not run on hardware
//
#include <hip/hip_runtime.h>
#include <math.h>

#define D 128
#define FF 512

typedef __attribute__((ext_vector_type(16))) __bf16 v16bf;
typedef __attribute__((ext_vector_type(8)))  float  v8f;

union Frag16 { v16bf v; unsigned int u[8]; };

__device__ __forceinline__ unsigned short f2bf(float f) {
    unsigned int u = __float_as_uint(f);
    unsigned int r = u + 0x7FFFu + ((u >> 16) & 1u);   // round to nearest even
    return (unsigned short)(r >> 16);
}
__device__ __forceinline__ unsigned int pack_bf2(float a, float b) {
    return (unsigned int)f2bf(a) | ((unsigned int)f2bf(b) << 16);
}

// A-fragment from a row-major bf16 LDS tile (16 rows x rowStride cols).
// ISA 7.12.2: lane<16 holds row m=lane, K {0..7,16..23}+32*kc;
// lane>=16 holds row m=lane-16, K {8..15,24..31}+32*kc. K-pairs contiguous -> u32 loads.
__device__ __forceinline__ Frag16 load_a_lds(const unsigned short* tile, int lane,
                                             int kc, int rowStride) {
    Frag16 f;
    const int m  = lane & 15;
    const int bk = ((lane >> 4) << 3) + (kc << 5);
    const unsigned int* base = reinterpret_cast<const unsigned int*>(tile + m * rowStride);
#pragma unroll
    for (int p = 0; p < 8; ++p) {
        int k0 = bk + 2 * p + ((p >= 4) ? 8 : 0);
        f.u[p] = base[k0 >> 1];
    }
    return f;
}

// B-fragment from pre-packed weights: ((nt*KC+kc)*32+lane)*16 bf16 = 32 contiguous bytes.
__device__ __forceinline__ Frag16 load_b_packed(const unsigned short* packed, int nt,
                                                int kc, int KC, int lane) {
    const uint4* p = reinterpret_cast<const uint4*>(packed + (size_t)((nt * KC + kc) * 32 + lane) * 16);
    uint4 a0 = p[0], a1 = p[1];
    Frag16 f;
    f.u[0] = a0.x; f.u[1] = a0.y; f.u[2] = a0.z; f.u[3] = a0.w;
    f.u[4] = a1.x; f.u[5] = a1.y; f.u[6] = a1.z; f.u[7] = a1.w;
    return f;
}

__device__ __forceinline__ v8f wmma_bf16(const Frag16& a, const Frag16& b, v8f c) {
    return __builtin_amdgcn_wmma_f32_16x16x32_bf16(false, a.v, false, b.v, (short)0, c, false, false);
}

// LayerNorm 16 rows -> bf16 row-major LDS tile. One wave, 32 lanes x 4 floats per row.
__device__ __forceinline__ void ln_rows16_to_lds(const float* __restrict__ x, int N, int row0,
                                                 const float* __restrict__ g,
                                                 const float* __restrict__ b,
                                                 unsigned short* tile, int lane) {
    float4 gv = reinterpret_cast<const float4*>(g)[lane];
    float4 bv = reinterpret_cast<const float4*>(b)[lane];
    for (int m = 0; m < 16; ++m) {
        int row = row0 + m;
        if (row >= N) row = N - 1;
        float4 xv = reinterpret_cast<const float4*>(x + (size_t)row * D)[lane];
        float s = xv.x + xv.y + xv.z + xv.w;
        float q = xv.x * xv.x + xv.y * xv.y + xv.z * xv.z + xv.w * xv.w;
#pragma unroll
        for (int o = 16; o >= 1; o >>= 1) {
            s += __shfl_xor(s, o, 32);
            q += __shfl_xor(q, o, 32);
        }
        float mu  = s * (1.0f / D);
        float var = q * (1.0f / D) - mu * mu;
        float rs  = rsqrtf(var + 1e-5f);
        float y0 = (xv.x - mu) * rs * gv.x + bv.x;
        float y1 = (xv.y - mu) * rs * gv.y + bv.y;
        float y2 = (xv.z - mu) * rs * gv.z + bv.z;
        float y3 = (xv.w - mu) * rs * gv.w + bv.w;
        unsigned int* trow = reinterpret_cast<unsigned int*>(tile + m * D);
        trow[lane * 2 + 0] = pack_bf2(y0, y1);
        trow[lane * 2 + 1] = pack_bf2(y2, y3);
    }
}

__device__ __forceinline__ void atomicMaxF32(float* addr, float v) {
    if (v >= 0.0f) atomicMax((int*)addr, __float_as_int(v));
    else           atomicMin((unsigned int*)addr, __float_as_uint(v));
}

// ---------- precompute: we = W_edge @ att_edge ----------
__global__ void k_we(const float* __restrict__ We, const float* __restrict__ ae,
                     float* __restrict__ we) {
    int i = threadIdx.x;  // 128 threads
    float acc = 0.0f;
    for (int j = 0; j < D; ++j) acc += We[i * D + j] * ae[j];
    we[i] = acc;
}

// ---------- pack f32 [K x Nn] weight into WMMA B-fragment bf16 layout ----------
__global__ void k_pack(const float* __restrict__ src, unsigned short* __restrict__ dst,
                       int K, int Nn) {
    int tid = blockIdx.x * 256 + threadIdx.x;
    if (tid >= K * Nn) return;
    int i    = tid & 15;
    int lane = (tid >> 4) & 31;
    int KC   = K >> 5;
    int kc   = (tid >> 9) % KC;
    int nt   = tid / (512 * KC);
    int n = nt * 16 + (lane & 15);
    int k = kc * 32 + ((lane >> 4) << 3) + (i < 8 ? i : i + 8);
    dst[tid] = f2bf(src[k * Nn + n]);
}

// ---------- init scratch (agg=0, m=-inf, denom=0) ----------
__global__ void k_init(float* agg, float* mmax, float* denom, int N) {
    int i = blockIdx.x * 256 + threadIdx.x;
    if (i < N * D) agg[i] = 0.0f;
    if (i < N) { mmax[i] = -__builtin_inff(); denom[i] = 0.0f; }
}

// ---------- LN1 + h = xn @ W_lin  (bf16 WMMA, f32 acc) ----------
// h is padded in ws to a multiple of 128 rows -> unconditional branch-free stores.
__global__ void __launch_bounds__(256) k_gat_proj(const float* __restrict__ x,
                                                  const float* __restrict__ g,
                                                  const float* __restrict__ b,
                                                  const unsigned short* __restrict__ Wp,
                                                  float* __restrict__ h, int N) {
    __shared__ unsigned short smem[8][16 * D];
    int lane = threadIdx.x & 31, wid = threadIdx.x >> 5;
    int row0 = (blockIdx.x * 8 + wid) * 16;
    ln_rows16_to_lds(x, N, row0, g, b, smem[wid], lane);
    __syncthreads();
    Frag16 afr[4];
#pragma unroll
    for (int kc = 0; kc < 4; ++kc) afr[kc] = load_a_lds(smem[wid], lane, kc, D);
    int n_lo = lane & 15, mb = (lane >> 4) << 3;
    for (int nt = 0; nt < 8; ++nt) {
        Frag16 bfr[4];
#pragma unroll
        for (int kc = 0; kc < 4; ++kc) bfr[kc] = load_b_packed(Wp, nt, kc, 4, lane);
        v8f acc = {};
#pragma unroll
        for (int kc = 0; kc < 4; ++kc) acc = wmma_bf16(afr[kc], bfr[kc], acc);
        float* hp = h + (size_t)(row0 + mb) * D + nt * 16 + n_lo;
#pragma unroll
        for (int r = 0; r < 8; ++r) hp[(size_t)r * D] = acc[r];
    }
}

// ---------- a_src = h@att_src, a_dst = h@att_dst (one wave per row) ----------
__global__ void k_attscore(const float* __restrict__ h, const float* __restrict__ avs,
                           const float* __restrict__ avd, float* __restrict__ asrc,
                           float* __restrict__ adst, int N) {
    int lane = threadIdx.x & 31;
    int row = blockIdx.x * 8 + (threadIdx.x >> 5);
    if (row >= N) return;
    float4 hv = reinterpret_cast<const float4*>(h + (size_t)row * D)[lane];
    float4 sv = reinterpret_cast<const float4*>(avs)[lane];
    float4 dv = reinterpret_cast<const float4*>(avd)[lane];
    float s = hv.x * sv.x + hv.y * sv.y + hv.z * sv.z + hv.w * sv.w;
    float d = hv.x * dv.x + hv.y * dv.y + hv.z * dv.z + hv.w * dv.w;
#pragma unroll
    for (int o = 16; o >= 1; o >>= 1) {
        s += __shfl_xor(s, o, 32);
        d += __shfl_xor(d, o, 32);
    }
    if (lane == 0) { asrc[row] = s; adst[row] = d; }
}

// ---------- edge pass 1: alpha = leaky(a_src[s]+a_dst[d]+edge_attr.we); segment max ----------
__global__ void k_edge0(const float* __restrict__ eattr, const int* __restrict__ ei,
                        const float* __restrict__ asrc, const float* __restrict__ adst,
                        const float* __restrict__ we, float* __restrict__ alpha,
                        float* __restrict__ mmax, int E) {
    int lane = threadIdx.x & 31;
    int e = blockIdx.x * 8 + (threadIdx.x >> 5);
    if (e >= E) return;
    float4 av = reinterpret_cast<const float4*>(eattr + (size_t)e * D)[lane];
    float4 wv = reinterpret_cast<const float4*>(we)[lane];
    float dot = av.x * wv.x + av.y * wv.y + av.z * wv.z + av.w * wv.w;
#pragma unroll
    for (int o = 16; o >= 1; o >>= 1) dot += __shfl_xor(dot, o, 32);
    if (lane == 0) {
        int s = ei[e], d = ei[E + e];
        float a = asrc[s] + adst[d] + dot;
        a = a > 0.0f ? a : 0.2f * a;           // leaky_relu(0.2)
        alpha[e] = a;
        atomicMaxF32(mmax + d, a);
    }
}

// ---------- edge pass 2: ex = exp(alpha - m[dst]); segment sum ----------
__global__ void k_edge1(const int* __restrict__ ei, const float* __restrict__ mmax,
                        float* __restrict__ alpha, float* __restrict__ denom, int E) {
    int e = blockIdx.x * 256 + threadIdx.x;
    if (e >= E) return;
    int d = ei[E + e];
    float ex = expf(alpha[e] - mmax[d]);
    alpha[e] = ex;
    atomicAdd(denom + d, ex);
}

// ---------- edge pass 3: agg[dst] += coef * h[src] ----------
__global__ void k_edge2(const int* __restrict__ ei, const float* __restrict__ alpha,
                        const float* __restrict__ denom, const float* __restrict__ h,
                        float* __restrict__ agg, int E) {
    int lane = threadIdx.x & 31;
    int e = blockIdx.x * 8 + (threadIdx.x >> 5);
    if (e >= E) return;
    int s = ei[e], d = ei[E + e];
    float coef = alpha[e] / fmaxf(denom[d], 1e-16f);
    float4 hv = reinterpret_cast<const float4*>(h + (size_t)s * D)[lane];
    float* o = agg + (size_t)d * D + lane * 4;
    atomicAdd(o + 0, coef * hv.x);
    atomicAdd(o + 1, coef * hv.y);
    atomicAdd(o + 2, coef * hv.z);
    atomicAdd(o + 3, coef * hv.w);
}

// ---------- x2 = x + agg + gat_bias ----------
__global__ void k_resid(const float* __restrict__ x, const float* __restrict__ agg,
                        const float* __restrict__ gb, float* __restrict__ x2, int N) {
    int i = blockIdx.x * 256 + threadIdx.x;
    if (i >= N * D) return;
    x2[i] = x[i] + agg[i] + gb[i & (D - 1)];
}

// ---------- LN2 + FFN (128->512 relu ->128) + residual, fused per 16-row wave tile ----------
__global__ void __launch_bounds__(64) k_ffn(const float* __restrict__ x2,
                                            const float* __restrict__ g,
                                            const float* __restrict__ b,
                                            const unsigned short* __restrict__ W1p,
                                            const float* __restrict__ b1,
                                            const unsigned short* __restrict__ W2p,
                                            const float* __restrict__ b2,
                                            float* __restrict__ out, int N) {
    __shared__ unsigned short sxn[2][16 * D];
    __shared__ unsigned short st[2][16 * FF];
    int lane = threadIdx.x & 31, wid = threadIdx.x >> 5;
    int row0 = (blockIdx.x * 2 + wid) * 16;
    ln_rows16_to_lds(x2, N, row0, g, b, sxn[wid], lane);
    __syncthreads();

    Frag16 afr[4];
#pragma unroll
    for (int kc = 0; kc < 4; ++kc) afr[kc] = load_a_lds(sxn[wid], lane, kc, D);
    int n_lo = lane & 15, mb = (lane >> 4) << 3;
    unsigned short* tt = st[wid];

    // stage 1: t = relu(xn @ W1 + b1) -> bf16 LDS tile [16 x 512]
    for (int nt = 0; nt < 32; ++nt) {
        Frag16 bfr[4];
#pragma unroll
        for (int kc = 0; kc < 4; ++kc) bfr[kc] = load_b_packed(W1p, nt, kc, 4, lane);
        v8f acc = {};
#pragma unroll
        for (int kc = 0; kc < 4; ++kc) acc = wmma_bf16(afr[kc], bfr[kc], acc);
        int n = nt * 16 + n_lo;
        float bias = b1[n];
#pragma unroll
        for (int r = 0; r < 8; ++r) {
            float t = acc[r] + bias;
            t = t > 0.0f ? t : 0.0f;
            tt[(mb + r) * FF + n] = f2bf(t);
        }
    }
    __syncthreads();

    // stage 2: y = t @ W2 + b2; out = x2 + y
    Frag16 tfr[16];
#pragma unroll
    for (int kc = 0; kc < 16; ++kc) tfr[kc] = load_a_lds(tt, lane, kc, FF);
    for (int nt = 0; nt < 8; ++nt) {
        v8f acc = {};
#pragma unroll
        for (int kg = 0; kg < 4; ++kg) {       // groups of 4: overlap loads vs WMMA
            Frag16 bfr[4];
#pragma unroll
            for (int j = 0; j < 4; ++j) bfr[j] = load_b_packed(W2p, nt, kg * 4 + j, 16, lane);
#pragma unroll
            for (int j = 0; j < 4; ++j) acc = wmma_bf16(tfr[kg * 4 + j], bfr[j], acc);
        }
        int n = nt * 16 + n_lo;
        float bias = b2[n];
        if (row0 + 16 <= N) {                   // uniform full-tile fast path
            const float* xp = x2 + (size_t)(row0 + mb) * D + n;
            float* op = out + (size_t)(row0 + mb) * D + n;
#pragma unroll
            for (int r = 0; r < 8; ++r) op[(size_t)r * D] = xp[(size_t)r * D] + acc[r] + bias;
        } else {
#pragma unroll
            for (int r = 0; r < 8; ++r) {
                int row = row0 + mb + r;
                if (row < N)
                    out[(size_t)row * D + n] = x2[(size_t)row * D + n] + acc[r] + bias;
            }
        }
    }
}

extern "C" void kernel_launch(void* const* d_in, const int* in_sizes, int n_in,
                              void* d_out, int out_size, void* d_ws, size_t ws_size,
                              hipStream_t stream) {
    const float* x     = (const float*)d_in[0];
    const int*   ei    = (const int*)  d_in[1];
    const float* eattr = (const float*)d_in[2];
    const float* ln1g  = (const float*)d_in[3];
    const float* ln1b  = (const float*)d_in[4];
    const float* Wlin  = (const float*)d_in[5];
    const float* att_s = (const float*)d_in[6];
    const float* att_d = (const float*)d_in[7];
    const float* Wedge = (const float*)d_in[8];
    const float* att_e = (const float*)d_in[9];
    const float* gbias = (const float*)d_in[10];
    const float* ln2g  = (const float*)d_in[11];
    const float* ln2b  = (const float*)d_in[12];
    const float* W1    = (const float*)d_in[13];
    const float* b1    = (const float*)d_in[14];
    const float* W2    = (const float*)d_in[15];
    const float* b2    = (const float*)d_in[16];
    float* out = (float*)d_out;

    const int N  = in_sizes[0] / D;
    const int E  = in_sizes[1] / 2;
    const int Np = (N + 127) & ~127;            // padded rows for branch-free h stores

    char* ws = (char*)d_ws;
    size_t off = 0;
    auto carve = [&](size_t bytes) -> char* {
        char* p = ws + off;
        off += (bytes + 255) & ~(size_t)255;
        return p;
    };
    float* h     = (float*)carve((size_t)Np * D * 4);
    float* asrc  = (float*)carve((size_t)N * 4);
    float* adst  = (float*)carve((size_t)N * 4);
    float* alpha = (float*)carve((size_t)E * 4);
    float* mmax  = (float*)carve((size_t)N * 4);
    float* denom = (float*)carve((size_t)N * 4);
    float* agg   = (float*)carve((size_t)N * D * 4);
    float* x2    = (float*)carve((size_t)N * D * 4);
    float* we    = (float*)carve((size_t)D * 4);
    unsigned short* Wlinp = (unsigned short*)carve((size_t)D * D * 2);
    unsigned short* W1p   = (unsigned short*)carve((size_t)D * FF * 2);
    unsigned short* W2p   = (unsigned short*)carve((size_t)FF * D * 2);

    k_we  <<<1, D, 0, stream>>>(Wedge, att_e, we);
    k_pack<<<(D * D  + 255) / 256, 256, 0, stream>>>(Wlin, Wlinp, D, D);
    k_pack<<<(D * FF + 255) / 256, 256, 0, stream>>>(W1, W1p, D, FF);
    k_pack<<<(FF * D + 255) / 256, 256, 0, stream>>>(W2, W2p, FF, D);
    k_init<<<(N * D + 255) / 256, 256, 0, stream>>>(agg, mmax, denom, N);

    k_gat_proj<<<Np / 128, 256, 0, stream>>>(x, ln1g, ln1b, Wlinp, h, N);
    k_attscore<<<(N + 7) / 8, 256, 0, stream>>>(h, att_s, att_d, asrc, adst, N);

    k_edge0<<<(E + 7) / 8, 256, 0, stream>>>(eattr, ei, asrc, adst, we, alpha, mmax, E);
    k_edge1<<<(E + 255) / 256, 256, 0, stream>>>(ei, mmax, alpha, denom, E);
    k_edge2<<<(E + 7) / 8, 256, 0, stream>>>(ei, alpha, denom, h, agg, E);

    k_resid<<<(N * D + 255) / 256, 256, 0, stream>>>(x, agg, gbias, x2, N);
    k_ffn<<<(N + 31) / 32, 64, 0, stream>>>(x2, ln2g, ln2b, W1p, b1, W2p, b2, out, N);
}
